// LSTMModel_57166014709915
// MI455X (gfx1250) — compile-verified, matrix operand order checked by reading, not run on hardware
//
#include <hip/hip_runtime.h>
#include <hip/hip_bf16.h>

// ---------------------------------------------------------------------------
// Bidirectional 2-layer LSTM (B=32, T=512, I=256, H=512) + FC head for
// MI455X / gfx1250.  All GEMMs use v_wmma_f32_16x16x32_bf16 (bf16 in, fp32
// accumulate).  The sequential recurrence runs as a persistent kernel:
// 16 WGs per direction, each holding its 128 rows of W_hh in LDS (130KB of
// the 320KB/WGP), cell state resident in WMMA accumulator registers, and a
// per-step inter-WG atomic barrier.
// ---------------------------------------------------------------------------

typedef __bf16        v16bf  __attribute__((ext_vector_type(16)));
typedef float         v8f    __attribute__((ext_vector_type(8)));
typedef unsigned int  uint4v __attribute__((ext_vector_type(4)));

#define TT  512
#define BB  32
#define II  256
#define HH  512
#define OO  256
#define G4H (4 * HH)

static constexpr int    CHUNK      = 32;            // hidden units per WG
static constexpr int    NWG_DIR    = HH / CHUNK;    // 16 WGs per direction
static constexpr int    SPITCH     = 520;           // bf16 elems per LDS row (1040B, 16B aligned, bank-conflict free)
static constexpr size_t SMEM_BYTES = (size_t)4 * CHUNK * SPITCH * sizeof(__bf16); // 133120

// Load one 16x32 bf16 WMMA fragment (A or B).  Per the CDNA5 ISA layout,
// lane L holds row (L%16); element e sits at K = (e/8)*16 + (L/16)*8 + e%8,
// i.e. two contiguous 16-byte chunks at +0 and +32 bytes from the lane base.
__device__ __forceinline__ v16bf ld_frag(const __bf16* p) {
  union { v16bf v; uint4v q[2]; } f;
  f.q[0] = *(const uint4v*)(p);
  f.q[1] = *(const uint4v*)(p + 16);
  return f.v;
}

__device__ __forceinline__ v8f wmma_bf16(v16bf a, v16bf b, v8f c) {
  // (neg_a, A, neg_b, B, c_mod, C, reuse_a, reuse_b)
  return __builtin_amdgcn_wmma_f32_16x16x32_bf16(false, a, false, b, (short)0, c, false, false);
}

__device__ __forceinline__ float fsigmoid(float x) { return 1.0f / (1.0f + __expf(-x)); }
__device__ __forceinline__ float ftanh_fast(float x) { return 1.0f - 2.0f / (__expf(2.0f * x) + 1.0f); }

// ---------------------------------------------------------------------------
// Persistent LSTM recurrence.  grid = 2 directions * 16 WGs, 128 threads/WG.
// Each WG owns 32 hidden units (all 4 gates).  Wave v: M-tile m=v/2 (batch
// rows 16m..16m+15), column half u=v%2.  gates = Xp[t] + h_{t-1} @ W_hh^T + b.
// Cell state stays in accumulator-layout registers across all 512 steps.
// ---------------------------------------------------------------------------
template <bool WRITE_OUT>
__global__ __launch_bounds__(128) void lstm_rec_kernel(
    const float* __restrict__ Xp_f, const float* __restrict__ Xp_b, int rev_b,
    const __bf16* __restrict__ Whh,   // [4H][H] bf16 row-major
    const float*  __restrict__ bias,  // [4H]  (b_ih + b_hh)
    __bf16* Hf, __bf16* Hb,           // [(T+1)][B][H] bf16, slot 0 zeroed
    float* gates_out, float* c_out,   // layer-1 only
    unsigned int* bar_base)           // two counters (dir*16 apart)
{
  extern __shared__ __bf16 sW[];      // [4*CHUNK rows][SPITCH]

  const int dir = blockIdx.x / NWG_DIR;
  const int wg  = blockIdx.x % NWG_DIR;
  const float* __restrict__ Xp = dir ? Xp_b : Xp_f;
  __bf16* Hbuf = dir ? Hb : Hf;
  unsigned int* bar = bar_base + dir * 16;
  const int rev = dir ? rev_b : 0;

  // Stage this WG's 128 W_hh rows into LDS (row = g*CHUNK + jl, padded pitch).
  for (int idx = threadIdx.x; idx < 4 * CHUNK * (HH / 8); idx += blockDim.x) {
    const int lrow = idx / (HH / 8);
    const int kc   = idx % (HH / 8);
    const int g  = lrow / CHUNK;
    const int jl = lrow % CHUNK;
    const __bf16* src = Whh + (size_t)(g * HH + wg * CHUNK + jl) * HH + kc * 8;
    *(uint4v*)(sW + (size_t)lrow * SPITCH + kc * 8) = *(const uint4v*)src;
  }
  __syncthreads();

  const int wave = threadIdx.x >> 5;
  const int lane = threadIdx.x & 31;
  const int r    = lane & 15;
  const int half = lane >> 4;
  const int m    = wave >> 1;          // batch-row tile
  const int u    = wave & 1;           // 16-wide column sub-block
  const int jglob = wg * CHUNK + u * 16 + r;   // this lane's hidden unit

  const float bI = bias[0 * HH + jglob];
  const float bF = bias[1 * HH + jglob];
  const float bG = bias[2 * HH + jglob];
  const float bO = bias[3 * HH + jglob];

  // Time-invariant B-fragment lane bases (one per gate) in LDS.
  const __bf16* sB0 = sW + (size_t)(0 * CHUNK + u * 16 + r) * SPITCH + half * 8;
  const __bf16* sB1 = sW + (size_t)(1 * CHUNK + u * 16 + r) * SPITCH + half * 8;
  const __bf16* sB2 = sW + (size_t)(2 * CHUNK + u * 16 + r) * SPITCH + half * 8;
  const __bf16* sB3 = sW + (size_t)(3 * CHUNK + u * 16 + r) * SPITCH + half * 8;

  v8f c_reg = {};   // cell state: 8 batch rows per lane, resident all 512 steps

  for (int t = 0; t < TT; ++t) {
    const int xt = rev ? (TT - 1 - t) : t;
    const float* __restrict__ xp = Xp + (size_t)xt * BB * G4H;
    const __bf16* hprev = Hbuf + (size_t)t * BB * HH + (size_t)(m * 16 + r) * HH + half * 8;

    v8f aI = {}, aF = {}, aG = {}, aO = {};
#pragma unroll
    for (int kt = 0; kt < HH / 32; ++kt) {
      v16bf a = ld_frag(hprev + kt * 32);
      aI = wmma_bf16(a, ld_frag(sB0 + kt * 32), aI);
      aF = wmma_bf16(a, ld_frag(sB1 + kt * 32), aF);
      aG = wmma_bf16(a, ld_frag(sB2 + kt * 32), aG);
      aO = wmma_bf16(a, ld_frag(sB3 + kt * 32), aO);
    }

    __bf16* hout = Hbuf + (size_t)(t + 1) * BB * HH;
#pragma unroll
    for (int vr = 0; vr < 8; ++vr) {
      const int brow = m * 16 + half * 8 + vr;          // C/D layout row
      const float* xr = xp + (size_t)brow * G4H;
      float gi = aI[vr] + xr[0 * HH + jglob] + bI;
      float gf = aF[vr] + xr[1 * HH + jglob] + bF;
      float gg = aG[vr] + xr[2 * HH + jglob] + bG;
      float go = aO[vr] + xr[3 * HH + jglob] + bO;
      float si = fsigmoid(gi);
      float sf = fsigmoid(gf);
      float tg = ftanh_fast(gg);
      float so = fsigmoid(go);
      float c  = sf * c_reg[vr] + si * tg;
      c_reg[vr] = c;
      float h = so * ftanh_fast(c);
      hout[(size_t)brow * HH + jglob] = (__bf16)h;
      if (WRITE_OUT) {
        const size_t col     = (size_t)dir * HH + jglob;
        const size_t kstride = (size_t)BB * (2 * HH);
        const size_t gbase   = (((size_t)t * 4) * BB + brow) * (2 * HH) + col;
        __builtin_nontemporal_store(si, &gates_out[gbase + 0 * kstride]);
        __builtin_nontemporal_store(sf, &gates_out[gbase + 1 * kstride]);
        __builtin_nontemporal_store(tg, &gates_out[gbase + 2 * kstride]);
        __builtin_nontemporal_store(so, &gates_out[gbase + 3 * kstride]);
        __builtin_nontemporal_store(c,  &c_out[((size_t)t * BB + brow) * (2 * HH) + col]);
      }
    }

    // Inter-WG barrier within this direction (16 WGs, all co-resident).
    __threadfence();          // release this WG's h stores
    __syncthreads();
    if (threadIdx.x == 0) {
      __hip_atomic_fetch_add(bar, 1u, __ATOMIC_RELEASE, __HIP_MEMORY_SCOPE_AGENT);
      const unsigned int target = (unsigned int)(NWG_DIR * (t + 1));
      while (__hip_atomic_load(bar, __ATOMIC_ACQUIRE, __HIP_MEMORY_SCOPE_AGENT) < target) {
        __builtin_amdgcn_s_sleep(2);
      }
    }
    __syncthreads();
    __threadfence();          // acquire: make peers' h visible to all waves
  }
}

// ---------------------------------------------------------------------------
// C[M,N] = A[M,K] @ W[N,K]^T ; A,W bf16 row-major, C fp32.  128 threads,
// 4 waves, each wave a 32x32 register tile -> 64x64 per block.
// ---------------------------------------------------------------------------
__global__ __launch_bounds__(128) void gemm_xw_kernel(
    const __bf16* __restrict__ A, const __bf16* __restrict__ W,
    float* __restrict__ C, int M, int N, int K)
{
  const int wave = threadIdx.x >> 5, lane = threadIdx.x & 31;
  const int r = lane & 15, half = lane >> 4;
  const int m0 = blockIdx.y * 64 + (wave >> 1) * 32;
  const int n0 = blockIdx.x * 64 + (wave & 1) * 32;
  (void)M;
  const __bf16* Ap0 = A + (size_t)(m0 + r) * K + half * 8;
  const __bf16* Ap1 = A + (size_t)(m0 + 16 + r) * K + half * 8;
  const __bf16* Wp0 = W + (size_t)(n0 + r) * K + half * 8;
  const __bf16* Wp1 = W + (size_t)(n0 + 16 + r) * K + half * 8;
  v8f a00 = {}, a01 = {}, a10 = {}, a11 = {};
  for (int k0 = 0; k0 < K; k0 += 32) {
    v16bf x0 = ld_frag(Ap0 + k0);
    v16bf x1 = ld_frag(Ap1 + k0);
    v16bf w0 = ld_frag(Wp0 + k0);
    v16bf w1 = ld_frag(Wp1 + k0);
    a00 = wmma_bf16(x0, w0, a00);
    a01 = wmma_bf16(x0, w1, a01);
    a10 = wmma_bf16(x1, w0, a10);
    a11 = wmma_bf16(x1, w1, a11);
  }
#pragma unroll
  for (int vr = 0; vr < 8; ++vr) {
    const int row0 = m0 + half * 8 + vr;
    const int row1 = row0 + 16;
    C[(size_t)row0 * N + n0 + r]      = a00[vr];
    C[(size_t)row0 * N + n0 + 16 + r] = a01[vr];
    C[(size_t)row1 * N + n0 + r]      = a10[vr];
    C[(size_t)row1 * N + n0 + 16 + r] = a11[vr];
  }
}

// ---------------------------------------------------------------------------
// FC head: y[b,t,o] = [h1f(t) , h1b(t)] @ fc_w^T + fc_b.
// Af/Ab already point at slot 1 so row (t*B+b) is plain row-major [T*B, H].
// ---------------------------------------------------------------------------
__global__ __launch_bounds__(128) void fc_kernel(
    const __bf16* __restrict__ Af, const __bf16* __restrict__ Ab,
    const __bf16* __restrict__ Wfc,   // [O][2H] bf16 row-major
    const float* __restrict__ fcb, float* __restrict__ y)
{
  const int K = 2 * HH, N = OO;
  const int wave = threadIdx.x >> 5, lane = threadIdx.x & 31;
  const int r = lane & 15, half = lane >> 4;
  const int m0 = blockIdx.y * 64 + (wave >> 1) * 32;
  const int n0 = blockIdx.x * 64 + (wave & 1) * 32;
  v8f a00 = {}, a01 = {}, a10 = {}, a11 = {};
  for (int k0 = 0; k0 < K; k0 += 32) {
    const __bf16* Asrc = (k0 < HH) ? Af : Ab;
    const int kk = (k0 < HH) ? k0 : (k0 - HH);
    v16bf x0 = ld_frag(Asrc + (size_t)(m0 + r) * HH + kk + half * 8);
    v16bf x1 = ld_frag(Asrc + (size_t)(m0 + 16 + r) * HH + kk + half * 8);
    v16bf w0 = ld_frag(Wfc + (size_t)(n0 + r) * K + k0 + half * 8);
    v16bf w1 = ld_frag(Wfc + (size_t)(n0 + 16 + r) * K + k0 + half * 8);
    a00 = wmma_bf16(x0, w0, a00);
    a01 = wmma_bf16(x0, w1, a01);
    a10 = wmma_bf16(x1, w0, a10);
    a11 = wmma_bf16(x1, w1, a11);
  }
#pragma unroll
  for (int vr = 0; vr < 8; ++vr) {
    const int row0 = m0 + half * 8 + vr;   // row = t*B + b
    const int row1 = row0 + 16;
    const int t0 = row0 >> 5, b0 = row0 & 31;
    const int t1 = row1 >> 5, b1 = row1 & 31;
    const int oa = n0 + r, ob = n0 + 16 + r;
    y[((size_t)b0 * TT + t0) * OO + oa] = a00[vr] + fcb[oa];
    y[((size_t)b0 * TT + t0) * OO + ob] = a01[vr] + fcb[ob];
    y[((size_t)b1 * TT + t1) * OO + oa] = a10[vr] + fcb[oa];
    y[((size_t)b1 * TT + t1) * OO + ob] = a11[vr] + fcb[ob];
  }
}

// ------------------------------ utilities ----------------------------------
__global__ void f32_to_bf16_kernel(const float* __restrict__ in, __bf16* __restrict__ out, int n) {
  int i = blockIdx.x * blockDim.x + threadIdx.x;
  if (i < n) out[i] = (__bf16)in[i];
}

__global__ void transpose_x_kernel(const float* __restrict__ X, __bf16* __restrict__ xs) {
  int idx = blockIdx.x * blockDim.x + threadIdx.x;     // over T*B*I (out-linear)
  if (idx >= TT * BB * II) return;
  const int i = idx % II;
  const int tb = idx / II;
  const int b = tb % BB;
  const int t = tb / BB;
  xs[idx] = (__bf16)X[((size_t)b * TT + t) * II + i];
}

__global__ void add_bias_kernel(const float* __restrict__ a, const float* __restrict__ b,
                                float* __restrict__ o, int n) {
  int i = blockIdx.x * blockDim.x + threadIdx.x;
  if (i < n) o[i] = a[i] + b[i];
}

// ---------------------------------------------------------------------------
extern "C" void kernel_launch(void* const* d_in, const int* in_sizes, int n_in,
                              void* d_out, int out_size, void* d_ws, size_t ws_size,
                              hipStream_t stream)
{
  (void)in_sizes; (void)n_in; (void)out_size; (void)ws_size;

  const float* X        = (const float*)d_in[0];
  const float* w_ih_in  = (const float*)d_in[1];
  const float* w_hh_in  = (const float*)d_in[2];
  const float* b_ih_in  = (const float*)d_in[3];
  const float* b_hh_in  = (const float*)d_in[4];
  const float* w_ih_mid = (const float*)d_in[5];
  const float* w_hh_mid = (const float*)d_in[6];
  const float* b_ih_mid = (const float*)d_in[7];
  const float* b_hh_mid = (const float*)d_in[8];
  const float* fc_w     = (const float*)d_in[9];
  const float* fc_b     = (const float*)d_in[10];

  // ---- workspace layout (~352 MB) ----
  char* ws = (char*)d_ws;
  size_t off = 0;
  auto alloc = [&](size_t bytes) -> void* {
    void* p = ws + off;
    off = (off + bytes + 255) & ~(size_t)255;
    return p;
  };
  unsigned int* counters = (unsigned int*)alloc(256);               // 64 u32
  __bf16* xs_bf   = (__bf16*)alloc((size_t)TT * BB * II * 2);
  __bf16* wih0_bf = (__bf16*)alloc((size_t)G4H * II * 2);
  __bf16* whh0_bf = (__bf16*)alloc((size_t)G4H * HH * 2);
  __bf16* wih1_bf = (__bf16*)alloc((size_t)G4H * HH * 2);
  __bf16* whh1_bf = (__bf16*)alloc((size_t)G4H * HH * 2);
  __bf16* fcw_bf  = (__bf16*)alloc((size_t)OO * (2 * HH) * 2);
  float*  bias0   = (float*)alloc((size_t)G4H * 4);
  float*  bias1   = (float*)alloc((size_t)G4H * 4);
  float*  Xp0     = (float*)alloc((size_t)TT * BB * G4H * 4);       // reused as Xp1f
  float*  Xp1b    = (float*)alloc((size_t)TT * BB * G4H * 4);
  __bf16* H0f     = (__bf16*)alloc((size_t)(TT + 1) * BB * HH * 2);
  __bf16* H0b     = (__bf16*)alloc((size_t)(TT + 1) * BB * HH * 2);
  __bf16* H1f     = (__bf16*)alloc((size_t)(TT + 1) * BB * HH * 2);
  __bf16* H1b     = (__bf16*)alloc((size_t)(TT + 1) * BB * HH * 2);

  float* y_out     = (float*)d_out;                                  // [B,T,O]
  float* gates_out = y_out + (size_t)BB * TT * OO;                   // [T,4,B,2H]
  float* hn_out    = gates_out + (size_t)TT * 4 * BB * 2 * HH;       // [T,B,2H]

  // Allow >64KB dynamic LDS for the recurrence kernels (idempotent).
  (void)hipFuncSetAttribute(reinterpret_cast<const void*>(lstm_rec_kernel<false>),
                            hipFuncAttributeMaxDynamicSharedMemorySize, (int)SMEM_BYTES);
  (void)hipFuncSetAttribute(reinterpret_cast<const void*>(lstm_rec_kernel<true>),
                            hipFuncAttributeMaxDynamicSharedMemorySize, (int)SMEM_BYTES);

  // Re-init mutable state every call (d_ws is poisoned, not re-poisoned).
  (void)hipMemsetAsync(counters, 0, 256, stream);
  (void)hipMemsetAsync(H0f, 0, (size_t)BB * HH * 2, stream);   // h_{-1} = 0
  (void)hipMemsetAsync(H0b, 0, (size_t)BB * HH * 2, stream);
  (void)hipMemsetAsync(H1f, 0, (size_t)BB * HH * 2, stream);
  (void)hipMemsetAsync(H1b, 0, (size_t)BB * HH * 2, stream);

  // ---- precision conversion / prep ----
  int n;
  n = G4H * II;     f32_to_bf16_kernel<<<(n + 255) / 256, 256, 0, stream>>>(w_ih_in,  wih0_bf, n);
  n = G4H * HH;     f32_to_bf16_kernel<<<(n + 255) / 256, 256, 0, stream>>>(w_hh_in,  whh0_bf, n);
  n = G4H * HH;     f32_to_bf16_kernel<<<(n + 255) / 256, 256, 0, stream>>>(w_ih_mid, wih1_bf, n);
  n = G4H * HH;     f32_to_bf16_kernel<<<(n + 255) / 256, 256, 0, stream>>>(w_hh_mid, whh1_bf, n);
  n = OO * 2 * HH;  f32_to_bf16_kernel<<<(n + 255) / 256, 256, 0, stream>>>(fc_w,     fcw_bf,  n);
  add_bias_kernel<<<(G4H + 255) / 256, 256, 0, stream>>>(b_ih_in,  b_hh_in,  bias0, G4H);
  add_bias_kernel<<<(G4H + 255) / 256, 256, 0, stream>>>(b_ih_mid, b_hh_mid, bias1, G4H);
  n = TT * BB * II; transpose_x_kernel<<<(n + 255) / 256, 256, 0, stream>>>(X, xs_bf);

  const dim3 blk(128);
  const int MM = TT * BB;   // 16384

  // ---- layer 0 input projection: Xp0 = xs @ W_ih_in^T  (K=256, N=2048) ----
  gemm_xw_kernel<<<dim3(G4H / 64, MM / 64), blk, 0, stream>>>(xs_bf, wih0_bf, Xp0, MM, G4H, II);

  // ---- layer 0 recurrence (both directions; backward reads Xp0 reversed) --
  lstm_rec_kernel<false><<<dim3(2 * NWG_DIR), blk, SMEM_BYTES, stream>>>(
      Xp0, Xp0, /*rev_b=*/1, whh0_bf, bias0, H0f, H0b, nullptr, nullptr, counters);

  // ---- layer 1 input projections (K=512): reuse Xp0 buffer for forward ----
  gemm_xw_kernel<<<dim3(G4H / 64, MM / 64), blk, 0, stream>>>(H0f + (size_t)BB * HH, wih1_bf, Xp0,  MM, G4H, HH);
  gemm_xw_kernel<<<dim3(G4H / 64, MM / 64), blk, 0, stream>>>(H0b + (size_t)BB * HH, wih1_bf, Xp1b, MM, G4H, HH);

  // ---- layer 1 recurrence; emits gates + cells straight into d_out --------
  lstm_rec_kernel<true><<<dim3(2 * NWG_DIR), blk, SMEM_BYTES, stream>>>(
      Xp0, Xp1b, /*rev_b=*/0, whh1_bf, bias1, H1f, H1b, gates_out, hn_out, counters + 32);

  // ---- FC head + [T,B]->[B,T] transpose -----------------------------------
  fc_kernel<<<dim3(OO / 64, MM / 64), blk, 0, stream>>>(
      H1f + (size_t)BB * HH, H1b + (size_t)BB * HH, fcw_bf, fc_b, y_out);
}
